// PointNetSAModuleMSG_60610578481760
// MI455X (gfx1250) — matrix operand order repaired and checked by reading, not compiled, not run on hardware
//
#include <hip/hip_runtime.h>

// ---------------------------------------------------------------------------
// PointNet++ SA-MSG for MI455X (gfx1250, wave32, WMMA)
//   Kernel 1: farthest point sampling  -> query_xyz (written into d_out)
//   Kernel 2: fused ball-query + shared MLP (bf16 WMMA, f32 accum) + maxpool
// ---------------------------------------------------------------------------

typedef __attribute__((ext_vector_type(16))) __bf16 v16bf;
typedef __attribute__((ext_vector_type(8)))  __bf16 v8bf;
typedef __attribute__((ext_vector_type(8)))  float  v8f;

union AU { v16bf v; v8bf h[2]; };

#define Bn    8
#define Nn    8192
#define Cn    64
#define Sn    2048
#define SSTR  104          // staging row stride (bf16): 208B, conflict-free for 16-row frag loads
#define WAVES 2
#define MSGT  (WAVES * 32)
#define MSG_BLOCKS 512

// native f32 -> bf16 convert (hardware v_cvt on gfx1250)
__device__ __forceinline__ __bf16 f2bf(float f) { return (__bf16)f; }

__device__ __forceinline__ void lds_sync() {
  asm volatile("s_wait_dscnt 0" ::: "memory");
}

// ---------------------------------------------------------------------------
// Farthest point sampling: one block per batch, 1024 threads, 8 pts/thread.
// Emits selected point COORDS directly (query_xyz), matching reference order
// (first emitted point is index 0).
// ---------------------------------------------------------------------------
#define FPS_T 1024
#define FPS_PPT (Nn / FPS_T)

__global__ __launch_bounds__(FPS_T) void fps_kernel(const float* __restrict__ xyz,
                                                    float* __restrict__ qxyz) {
  int b = blockIdx.x;
  const float* px = xyz + (size_t)b * Nn * 3;
  float* q = qxyz + (size_t)b * Sn * 3;
  int tid = threadIdx.x, lane = tid & 31, wid = tid >> 5;

  float X[FPS_PPT], Y[FPS_PPT], Z[FPS_PPT], D[FPS_PPT];
  int base = tid * FPS_PPT;
#pragma unroll
  for (int i = 0; i < FPS_PPT; ++i) {
    X[i] = px[(base + i) * 3 + 0];
    Y[i] = px[(base + i) * 3 + 1];
    Z[i] = px[(base + i) * 3 + 2];
    D[i] = 1e10f;
  }

  __shared__ float sb[32];
  __shared__ int   si[32];
  __shared__ float sl[3];

  float lx = px[0], ly = px[1], lz = px[2];

  for (int it = 0; it < Sn; ++it) {
    if (tid == 0) { q[it * 3 + 0] = lx; q[it * 3 + 1] = ly; q[it * 3 + 2] = lz; }

    float best = -1.f; int bi = 0;
#pragma unroll
    for (int i = 0; i < FPS_PPT; ++i) {
      float dx = X[i] - lx, dy = Y[i] - ly, dz = Z[i] - lz;
      float d = dx * dx + dy * dy + dz * dz;
      D[i] = fminf(D[i], d);
      if (D[i] > best) { best = D[i]; bi = base + i; }
    }
#pragma unroll
    for (int off = 16; off > 0; off >>= 1) {
      float ob = __shfl_down(best, off, 32);
      int   oi = __shfl_down(bi,   off, 32);
      if (ob > best || (ob == best && oi < bi)) { best = ob; bi = oi; }
    }
    if (lane == 0) { sb[wid] = best; si[wid] = bi; }
    __syncthreads();
    if (wid == 0) {
      float wb = sb[lane]; int wi = si[lane];
#pragma unroll
      for (int off = 16; off > 0; off >>= 1) {
        float ob = __shfl_down(wb, off, 32);
        int   oi = __shfl_down(wi, off, 32);
        if (ob > wb || (ob == wb && oi < wi)) { wb = ob; wi = oi; }
      }
      if (lane == 0) { sl[0] = px[wi * 3 + 0]; sl[1] = px[wi * 3 + 1]; sl[2] = px[wi * 3 + 2]; }
    }
    __syncthreads();
    lx = sl[0]; ly = sl[1]; lz = sl[2];
    __syncthreads();
  }
}

// ---------------------------------------------------------------------------
// WMMA helpers (16x16x32 bf16, f32 accumulate)
// A layout: row m = lane&15; K runs [h*8,h*8+8) and [16+h*8,...) (h = lane>>4)
// B layout: col n = lane&15; K runs [h*8,h*8+8) and [16+h*8,...) as in A (weights stored [n][kpad])
// ---------------------------------------------------------------------------
__device__ __forceinline__ void load_afrags(const __bf16* sw, int lane, int kchunks, v16bf* a) {
  int m = lane & 15, h = lane >> 4;
  const __bf16* rp = sw + m * SSTR;
  AU au;
  au.h[0] = *(const v8bf*)(rp + h * 8);
  au.h[1] = *(const v8bf*)(rp + 16 + h * 8);
  a[0] = au.v;
  if (kchunks > 1) {
    au.h[0] = *(const v8bf*)(rp + 32 + h * 8);
    au.h[1] = *(const v8bf*)(rp + 48 + h * 8);
    a[1] = au.v;
  }
  if (kchunks > 2) {
    au.h[0] = *(const v8bf*)(rp + 64 + h * 8);
    au.h[1] = *(const v8bf*)(rp + 80 + h * 8);
    a[2] = au.v;
  }
}

__device__ __forceinline__ v8f wmma_acc(const v16bf* a, const __bf16* wrow, int kchunks) {
  v8f c = {0.f, 0.f, 0.f, 0.f, 0.f, 0.f, 0.f, 0.f};
  AU bu;
  bu.h[0] = *(const v8bf*)(wrow);
  bu.h[1] = *(const v8bf*)(wrow + 16);
  c = __builtin_amdgcn_wmma_f32_16x16x32_bf16(false, a[0], false, bu.v, (short)0, c, false, false);
  if (kchunks > 1) {
    bu.h[0] = *(const v8bf*)(wrow + 32);
    bu.h[1] = *(const v8bf*)(wrow + 48);
    c = __builtin_amdgcn_wmma_f32_16x16x32_bf16(false, a[1], false, bu.v, (short)0, c, false, false);
  }
  if (kchunks > 2) {
    bu.h[0] = *(const v8bf*)(wrow + 64);
    bu.h[1] = *(const v8bf*)(wrow + 80);
    c = __builtin_amdgcn_wmma_f32_16x16x32_bf16(false, a[2], false, bu.v, (short)0, c, false, false);
  }
  return c;
}

// intermediate layer: bias + ReLU, write back to staging tile (bf16)
__device__ __forceinline__ void layer_stage(int lane, __bf16* sw, const __bf16* wt,
                                            const float* bias, int kchunks, int kstore,
                                            int ntiles) {
  int m = lane & 15, h = lane >> 4;
  v16bf a[3];
  load_afrags(sw, lane, kchunks, a);
  for (int nt = 0; nt < ntiles; ++nt) {
    const __bf16* wrow = wt + (size_t)(nt * 16 + m) * kstore + h * 8;
    v8f c = wmma_acc(a, wrow, kchunks);
    float bb = bias[nt * 16 + m];
    __bf16* op = sw + (h * 8) * SSTR + nt * 16 + m;
#pragma unroll
    for (int r = 0; r < 8; ++r) {
      float v = fmaxf(c[r] + bb, 0.f);        // bias + ReLU (v_max_num_f32)
      op[r * SSTR] = f2bf(v);
    }
  }
}

// final layer (128 out): bias + ReLU + running max over the 8 rows held per lane
__device__ __forceinline__ void layer_final(int lane, const __bf16* sw, const __bf16* wt,
                                            const float* bias, int kchunks, int kstore,
                                            float* fmax) {
  int m = lane & 15, h = lane >> 4;
  v16bf a[3];
  load_afrags(sw, lane, kchunks, a);
#pragma unroll
  for (int nt = 0; nt < 8; ++nt) {
    const __bf16* wrow = wt + (size_t)(nt * 16 + m) * kstore + h * 8;
    v8f c = wmma_acc(a, wrow, kchunks);
    float bb = bias[nt * 16 + m];
    float pm = 0.f;
#pragma unroll
    for (int r = 0; r < 8; ++r) {
      float v = fmaxf(c[r] + bb, 0.f);
      pm = fmaxf(pm, v);
    }
    fmax[nt] = fmaxf(fmax[nt], pm);
  }
}

// ---------------------------------------------------------------------------
// Fused ball-query + MLP + maxpool. One wave per query (grid-stride).
// Layer dims: 67(pad 96) -> 64 -> d1 -> 128, d1 in {64, 96}.
// ---------------------------------------------------------------------------
__global__ __launch_bounds__(MSGT) void msg_kernel(
    const float* __restrict__ xyz, const float* __restrict__ feats,
    const float* __restrict__ qxyz,
    const float* __restrict__ w0, const float* __restrict__ b0,
    const float* __restrict__ w1, const float* __restrict__ b1,
    const float* __restrict__ w2, const float* __restrict__ b2,
    float* __restrict__ outf, int K, float r2, int d1, int chan_off) {
  extern __shared__ char smem[];
  const int ks0 = 104;           // layer0 k-store stride: 96 (padded 67) + 8
  const int ks1 = 72;            // layer1: 64 + 8
  const int ks2 = d1 + 8;        // layer2: d1 + 8
  __bf16* wt0 = (__bf16*)smem;                       // [64][ks0]
  __bf16* wt1 = wt0 + 64 * ks0;                      // [d1][ks1]
  __bf16* wt2 = wt1 + (size_t)d1 * ks1;              // [128][ks2]
  float*  bsA = (float*)(wt2 + 128 * ks2);
  float *bs0 = bsA, *bs1 = bsA + 64, *bs2 = bs1 + d1;
  int* idxb = (int*)(bs2 + 128);                     // [WAVES][32]
  __bf16* stage = (__bf16*)(idxb + WAVES * 32);      // [WAVES][16][SSTR]

  int tid = threadIdx.x, lane = tid & 31, wid = tid >> 5;

  // ---- stage weights into LDS: bf16, transposed [n][k], k-padded ----
  for (int e = tid; e < 64 * ks0; e += MSGT) {
    int n = e / ks0, k = e - n * ks0;
    wt0[e] = (k < 67) ? f2bf(w0[k * 64 + n]) : f2bf(0.f);
  }
  for (int e = tid; e < d1 * ks1; e += MSGT) {
    int n = e / ks1, k = e - n * ks1;
    wt1[e] = (k < 64) ? f2bf(w1[k * d1 + n]) : f2bf(0.f);
  }
  for (int e = tid; e < 128 * ks2; e += MSGT) {
    int n = e / ks2, k = e - n * ks2;
    wt2[e] = (k < d1) ? f2bf(w2[k * 128 + n]) : f2bf(0.f);
  }
  for (int e = tid; e < 64;  e += MSGT) bs0[e] = b0[e];
  for (int e = tid; e < d1;  e += MSGT) bs1[e] = b1[e];
  for (int e = tid; e < 128; e += MSGT) bs2[e] = b2[e];
  __syncthreads();

  __bf16* sw   = stage + (size_t)wid * 16 * SSTR;
  int*    idxw = idxb + wid * 32;

  int gw = blockIdx.x * WAVES + wid;
  int nw = gridDim.x * WAVES;
  int kc2 = d1 >> 5;       // layer2 k-chunks (2 or 3)
  int nt1 = d1 >> 4;       // layer1 n-tiles (4 or 6)
  int mtiles = K >> 4;     // 1 or 2

  for (int q = gw; q < Bn * Sn; q += nw) {
    int b = q >> 11, s = q & (Sn - 1);
    const float* xb = xyz + (size_t)b * Nn * 3;
    const float* fb = feats + (size_t)b * Cn * Nn;
    const float* qp = qxyz + (size_t)q * 3;
    float qx = qp[0], qy = qp[1], qz = qp[2];

    // ---- ball query: first K in-ball indices (smallest index first) ----
    int cnt = 0;
    for (int basept = 0; basept < Nn && cnt < K; basept += 32) {
      int n = basept + lane;
      float dx = xb[n * 3 + 0] - qx;
      float dy = xb[n * 3 + 1] - qy;
      float dz = xb[n * 3 + 2] - qz;
      bool in = (dx * dx + dy * dy + dz * dz) <= r2;
      unsigned m32 = (unsigned)__ballot(in);
      int pos = cnt + __popc(m32 & ((1u << lane) - 1u));
      if (in && pos < K) idxw[pos] = n;
      cnt += __popc(m32);
    }
    if (cnt > K) cnt = K;
    lds_sync();
    int first = idxw[0];                       // query is a support point -> cnt >= 1
    for (int i = cnt + lane; i < K; i += 32) idxw[i] = first;
    lds_sync();

    float fmax[8];
#pragma unroll
    for (int i = 0; i < 8; ++i) fmax[i] = 0.f;  // post-ReLU values are >= 0

    for (int mt = 0; mt < mtiles; ++mt) {
      // ---- build input tile: rows=neighbors; cols 0..2 rel-xyz, 3..66 feat, 67..95 zero
      for (int e = lane; e < 16 * 96; e += 32) {
        int r = e / 96, c = e - r * 96;
        int p = idxw[mt * 16 + r];
        float v;
        if (c < 3) {
          float qv = (c == 0) ? qx : ((c == 1) ? qy : qz);
          v = xb[p * 3 + c] - qv;
        } else if (c < 67) {
          v = fb[(size_t)(c - 3) * Nn + p];
        } else {
          v = 0.f;
        }
        sw[r * SSTR + c] = f2bf(v);
      }
      lds_sync();
      layer_stage(lane, sw, wt0, bs0, 3, ks0, 4);       // 96  -> 64
      lds_sync();
      layer_stage(lane, sw, wt1, bs1, 2, ks1, nt1);     // 64  -> d1
      lds_sync();
      layer_final(lane, sw, wt2, bs2, kc2, ks2, fmax);  // d1  -> 128, pool rows
      lds_sync();
    }

    // combine row-halves (lanes L / L^16) and emit 128 channels
#pragma unroll
    for (int nt = 0; nt < 8; ++nt) {
      float v = fmax[nt];
      float o = __shfl_xor(v, 16, 32);
      v = fmaxf(v, o);
      if (lane < 16)
        outf[((size_t)b * 256 + chan_off + nt * 16 + lane) * Sn + s] = v;
    }
  }
}

static inline size_t msg_smem_bytes(int d1) {
  size_t wt    = (size_t)2 * (64 * 104 + (size_t)d1 * 72 + 128 * (size_t)(d1 + 8));
  size_t bias  = 4 * (size_t)(64 + d1 + 128);
  size_t idx   = 4 * (size_t)(WAVES * 32);
  size_t stage = (size_t)2 * (WAVES * 16 * SSTR);
  return wt + bias + idx + stage;   // 48896 B (d1=64) / 61824 B (d1=96)
}

extern "C" void kernel_launch(void* const* d_in, const int* in_sizes, int n_in,
                              void* d_out, int out_size, void* d_ws, size_t ws_size,
                              hipStream_t stream) {
  (void)in_sizes; (void)n_in; (void)out_size; (void)d_ws; (void)ws_size;

  const float* xyz   = (const float*)d_in[0];
  const float* feats = (const float*)d_in[1];
  const float* s0w0 = (const float*)d_in[2];
  const float* s0b0 = (const float*)d_in[3];
  const float* s0w1 = (const float*)d_in[4];
  const float* s0b1 = (const float*)d_in[5];
  const float* s0w2 = (const float*)d_in[6];
  const float* s0b2 = (const float*)d_in[7];
  const float* s1w0 = (const float*)d_in[8];
  const float* s1b0 = (const float*)d_in[9];
  const float* s1w1 = (const float*)d_in[10];
  const float* s1b1 = (const float*)d_in[11];
  const float* s1w2 = (const float*)d_in[12];
  const float* s1b2 = (const float*)d_in[13];

  float* out      = (float*)d_out;
  float* qxyz     = out;                               // (B, S, 3)
  float* feat_out = out + (size_t)Bn * Sn * 3;         // (B, 256, S)

  fps_kernel<<<Bn, FPS_T, 0, stream>>>(xyz, qxyz);

  msg_kernel<<<MSG_BLOCKS, MSGT, msg_smem_bytes(64), stream>>>(
      xyz, feats, qxyz, s0w0, s0b0, s0w1, s0b1, s0w2, s0b2,
      feat_out, 16, 0.01f, 64, 0);

  msg_kernel<<<MSG_BLOCKS, MSGT, msg_smem_bytes(96), stream>>>(
      xyz, feats, qxyz, s1w0, s1b0, s1w1, s1b1, s1w2, s1b2,
      feat_out, 32, 0.04f, 96, 128);
}
